// Residual_91319594647707
// MI455X (gfx1250) — matrix operand order repaired and checked by reading, not compiled, not run on hardware
//
#include <hip/hip_runtime.h>

// ---------------------------------------------------------------------------
// Residual-flow logdet estimator for MI455X (gfx1250, wave32, WMMA).
// B=8192, D=512, H=2048. All GEMMs in bf16 with f32 WMMA accumulation.
// Memory-bound workload: bf16 operands halve traffic and keep the entire
// power-series working set (~180MB) resident in the 192MB L2.
// Tile staging uses GLOBAL_LOAD_ASYNC_TO_LDS (ASYNCcnt) with double-buffered
// LDS so each tile's load latency is hidden under the previous tile's WMMAs.
// ---------------------------------------------------------------------------

typedef __attribute__((ext_vector_type(16))) __bf16 v16bf;
typedef __attribute__((ext_vector_type(8)))  float  v8f;

#if defined(__has_builtin)
#if __has_builtin(__builtin_amdgcn_global_load_async_to_lds_b128) && \
    __has_builtin(__builtin_amdgcn_s_wait_asynccnt)
#define HAVE_ASYNC_LDS 1
#endif
#endif
#ifndef HAVE_ASYNC_LDS
#define HAVE_ASYNC_LDS 0
#endif

#if HAVE_ASYNC_LDS
// Builtin signature (probe-confirmed via diagnostics): (v4i addrspace(1)*,
// v4i addrspace(3)*, imm offset, imm cpol).
typedef int v4i_ __attribute__((ext_vector_type(4)));
typedef __attribute__((address_space(1))) v4i_ g_v4i;   // global (__device__)
typedef __attribute__((address_space(3))) v4i_ l_v4i;   // LDS (__shared__)
__device__ __forceinline__ g_v4i* to_global(const void* p) {
    return (g_v4i*)(unsigned long long)p;               // flat == global bits
}
__device__ __forceinline__ l_v4i* to_lds(const void* p) {
    // ISA 10.2: generic LDS address low 32 bits are the DS offset.
    return (l_v4i*)(unsigned int)(unsigned long long)p;
}
#endif

union Frag {
    v16bf v;
    uint4 u[2];
};

// ------------------------------ small utility kernels ----------------------

__global__ void cvt_f32_bf16(const float* __restrict__ in, __bf16* __restrict__ out, int n) {
    int i = blockIdx.x * blockDim.x + threadIdx.x;
    if (i < n) out[i] = (__bf16)in[i];
}

// in: [R][C] f32 row-major  ->  out: [C][R] bf16 row-major
__global__ void transpose_f32_bf16(const float* __restrict__ in, __bf16* __restrict__ out,
                                   int R, int C) {
    int i = blockIdx.x * blockDim.x + threadIdx.x;
    if (i < R * C) {
        int r = i / C, c = i % C;
        out[(size_t)c * R + r] = (__bf16)in[i];
    }
}

__global__ void zero_f32(float* __restrict__ p, int n) {
    int i = blockIdx.x * blockDim.x + threadIdx.x;
    if (i < n) p[i] = 0.0f;
}

// logdet[row] += coeff * dot(w[row,:], eps[row,:]);  one wave per row.
__global__ void rowdot_acc(const __bf16* __restrict__ w, const float* __restrict__ eps,
                           float* __restrict__ logdet, float coeff, int K) {
    int row  = blockIdx.x * 8 + (threadIdx.x >> 5);
    int lane = threadIdx.x & 31;
    const __bf16* wr = w   + (size_t)row * K;
    const float*  er = eps + (size_t)row * K;
    float s = 0.0f;
    for (int k = lane; k < K; k += 32) s += (float)wr[k] * er[k];
#pragma unroll
    for (int off = 16; off > 0; off >>= 1) s += __shfl_down(s, off, 32);
    if (lane == 0) logdet[row] += coeff * s;
}

// ------------------------------ WMMA GEMM ----------------------------------
// C[M][N] = A[M][K] @ B (B given N-major as Bn[N][K], i.e. columns of the
// mathematical B are contiguous in K) with fused epilogues.
// Block tile 128x128, K-step 32, 256 threads = 8 waves; each wave owns a
// 32x64 sub-tile = 2x4 WMMA accumulators (v8f each). LDS double-buffered,
// filled by per-lane async b128 copies (ASYNCcnt).

enum { EPI_TANH = 0, EPI_ADDX = 1, EPI_MULD = 2, EPI_PLAIN = 3 };

template <int EPI>
__global__ __launch_bounds__(256)
void gemm_wmma_bf16(const __bf16* __restrict__ A,    // [M][K]
                    const __bf16* __restrict__ Bn,   // [N][K]
                    const float*  __restrict__ bias, // [N] (TANH / ADDX)
                    const __bf16* __restrict__ dmul, // [M][N] (MULD)
                    const float*  __restrict__ addx, // [M][N] (ADDX)
                    float*        __restrict__ outF, // [M][N] f32 (ADDX)
                    __bf16*       __restrict__ outB, // [M][N] bf16
                    __bf16*       __restrict__ outB2,// [M][N] bf16 (TANH deriv)
                    int M, int N, int K) {
    __shared__ alignas(16) __bf16 sA[2][128][40];   // +8 pad vs bank conflicts
    __shared__ alignas(16) __bf16 sB[2][128][40];

    const int m0   = blockIdx.y * 128;
    const int n0   = blockIdx.x * 128;
    const int tid  = threadIdx.x;
    const int lane = tid & 31;
    const int wave = tid >> 5;
    const int wm   = wave >> 1;   // 0..3 -> 32 rows each
    const int wn   = wave & 1;    // 0..1 -> 64 cols each
    const int l15  = lane & 15;
    const int kh   = lane >> 4;

    // Stage one 128x32 tile pair of A/Bn into LDS buffer `buf`.
    auto stage = [&](int buf, int k0) {
#pragma unroll
        for (int i = 0; i < 2; ++i) {
            int c   = tid + 256 * i;
            int row = c >> 2;
            int cc  = (c & 3) * 8;
#if HAVE_ASYNC_LDS
            __builtin_amdgcn_global_load_async_to_lds_b128(
                to_global(A + (size_t)(m0 + row) * K + k0 + cc),
                to_lds(&sA[buf][row][cc]), 0, 0);
            __builtin_amdgcn_global_load_async_to_lds_b128(
                to_global(Bn + (size_t)(n0 + row) * K + k0 + cc),
                to_lds(&sB[buf][row][cc]), 0, 0);
#else
            *(uint4*)&sA[buf][row][cc] = *(const uint4*)(A  + (size_t)(m0 + row) * K + k0 + cc);
            *(uint4*)&sB[buf][row][cc] = *(const uint4*)(Bn + (size_t)(n0 + row) * K + k0 + cc);
#endif
        }
    };

    v8f acc[2][4] = {};

    stage(0, 0);                         // prologue: tile 0 in flight
    int buf = 0;
    for (int k0 = 0; k0 < K; k0 += 32) {
#if HAVE_ASYNC_LDS
        __builtin_amdgcn_s_wait_asynccnt(0);   // our copies for `buf` landed
#endif
        __syncthreads();                       // everyone's copies landed

        // Issue next tile into the other buffer; latency hidden by the WMMAs
        // below. L2 prefetch runs one more tile ahead of the async engine.
        if (k0 + 32 < K) stage(buf ^ 1, k0 + 32);
        if (k0 + 64 < K) {
            __builtin_prefetch(A  + (size_t)(m0 + (tid >> 1)) * K + k0 + 64, 0, 1);
            __builtin_prefetch(Bn + (size_t)(n0 + (tid >> 1)) * K + k0 + 64, 0, 1);
        }

        // B fragments: lane n=l15 holds K = kh*16 .. kh*16+15 contiguous (32B).
        Frag bfr[4];
#pragma unroll
        for (int nt = 0; nt < 4; ++nt) {
            int bn = wn * 64 + nt * 16 + l15;
            bfr[nt].u[0] = *(const uint4*)&sB[buf][bn][kh * 16];
            bfr[nt].u[1] = *(const uint4*)&sB[buf][bn][kh * 16 + 8];
        }
        // A fragments: lane m=l15 holds K = kh*16 .. kh*16+15 contiguous (32B).
#pragma unroll
        for (int mt = 0; mt < 2; ++mt) {
            Frag af;
            int am = wm * 32 + mt * 16 + l15;
            af.u[0] = *(const uint4*)&sA[buf][am][kh * 16];
            af.u[1] = *(const uint4*)&sA[buf][am][kh * 16 + 8];
#pragma unroll
            for (int nt = 0; nt < 4; ++nt) {
                acc[mt][nt] = __builtin_amdgcn_wmma_f32_16x16x32_bf16(
                    false, af.v, false, bfr[nt].v, (short)0, acc[mt][nt], false, false);
            }
        }
        __syncthreads();                 // reads of `buf` done before its reuse
        buf ^= 1;
    }

    // Epilogue. C/D layout: VGPR r -> (M = r + 8*kh, N = l15) inside a 16x16 tile.
#pragma unroll
    for (int mt = 0; mt < 2; ++mt) {
#pragma unroll
        for (int nt = 0; nt < 4; ++nt) {
#pragma unroll
            for (int r = 0; r < 8; ++r) {
                int row = m0 + wm * 32 + mt * 16 + kh * 8 + r;
                int col = n0 + wn * 64 + nt * 16 + l15;
                size_t idx = (size_t)row * N + col;
                float v = acc[mt][nt][r];
                if (EPI == EPI_TANH) {
                    v += bias[col];
                    float t = tanhf(v);
                    outB[idx]  = (__bf16)t;
                    outB2[idx] = (__bf16)(1.0f - t * t);
                } else if (EPI == EPI_ADDX) {
                    outF[idx] = addx[idx] + v + bias[col];
                } else if (EPI == EPI_MULD) {
                    outB[idx] = (__bf16)(v * (float)dmul[idx]);
                } else { // EPI_PLAIN
                    outB[idx] = (__bf16)v;
                }
            }
        }
    }
}

// ------------------------------ driver -------------------------------------

extern "C" void kernel_launch(void* const* d_in, const int* in_sizes, int n_in,
                              void* d_out, int out_size, void* d_ws, size_t ws_size,
                              hipStream_t stream) {
    (void)in_sizes; (void)n_in; (void)out_size; (void)ws_size;
    const int B = 8192, D = 512, H = 2048;

    const float* x   = (const float*)d_in[0];
    const float* eps = (const float*)d_in[1];
    const float* W1  = (const float*)d_in[2];   // [D][H]
    const float* b1  = (const float*)d_in[3];
    const float* W2  = (const float*)d_in[4];   // [H][H]
    const float* b2  = (const float*)d_in[5];
    const float* W3  = (const float*)d_in[6];   // [H][D]
    const float* b3  = (const float*)d_in[7];

    float* z      = (float*)d_out;              // [B][D]
    float* logdet = z + (size_t)B * D;          // [B]

    // Workspace layout (168 MB total).
    char* ws = (char*)d_ws;
    const size_t MB = 1ull << 20;
    __bf16* W1T = (__bf16*)(ws +   0 * MB);     // [H][D]  fwd GEMM1 operand
    __bf16* W1b = (__bf16*)(ws +   2 * MB);     // [D][H]  bwd GEMM (w@W1^T)
    __bf16* W2T = (__bf16*)(ws +   4 * MB);     // [H][H]  fwd GEMM2
    __bf16* W2b = (__bf16*)(ws +  12 * MB);     // [H][H]  bwd GEMM (t@W2^T)
    __bf16* W3T = (__bf16*)(ws +  20 * MB);     // [D][H]  fwd GEMM3
    __bf16* W3b = (__bf16*)(ws +  22 * MB);     // [H][D]  bwd GEMM (w@W3^T)
    __bf16* xb  = (__bf16*)(ws +  24 * MB);     // [B][D]
    __bf16* d1  = (__bf16*)(ws +  32 * MB);     // [B][H]  1 - h1^2
    __bf16* d2  = (__bf16*)(ws +  64 * MB);     // [B][H]  1 - h2^2
    __bf16* h1  = (__bf16*)(ws +  96 * MB);     // [B][H], reused as t2 in bwd
    __bf16* h2  = (__bf16*)(ws + 128 * MB);     // [B][H], reused as t1 in bwd
    __bf16* wv  = (__bf16*)(ws + 160 * MB);     // [B][D]  current VJP cotangent

    auto cdiv = [](int a, int b) { return (a + b - 1) / b; };

    // One-time conversions: bf16 copies of weights in both layouts + inputs.
    cvt_f32_bf16      <<<cdiv(D * H, 256), 256, 0, stream>>>(W1, W1b, D * H);
    transpose_f32_bf16<<<cdiv(D * H, 256), 256, 0, stream>>>(W1, W1T, D, H);
    cvt_f32_bf16      <<<cdiv(H * H, 256), 256, 0, stream>>>(W2, W2b, H * H);
    transpose_f32_bf16<<<cdiv(H * H, 256), 256, 0, stream>>>(W2, W2T, H, H);
    cvt_f32_bf16      <<<cdiv(H * D, 256), 256, 0, stream>>>(W3, W3b, H * D);
    transpose_f32_bf16<<<cdiv(H * D, 256), 256, 0, stream>>>(W3, W3T, H, D);
    cvt_f32_bf16      <<<cdiv(B * D, 256), 256, 0, stream>>>(x,   xb, B * D);
    cvt_f32_bf16      <<<cdiv(B * D, 256), 256, 0, stream>>>(eps, wv, B * D);
    zero_f32          <<<cdiv(B,     256), 256, 0, stream>>>(logdet, B);

    dim3 blk(256);

    // Forward: h1 = tanh(x@W1+b1); h2 = tanh(h1@W2+b2); z = x + h2@W3 + b3.
    gemm_wmma_bf16<EPI_TANH><<<dim3(H / 128, B / 128), blk, 0, stream>>>(
        xb, W1T, b1, nullptr, nullptr, nullptr, h1, d1, B, H, D);
    gemm_wmma_bf16<EPI_TANH><<<dim3(H / 128, B / 128), blk, 0, stream>>>(
        h1, W2T, b2, nullptr, nullptr, nullptr, h2, d2, B, H, H);
    gemm_wmma_bf16<EPI_ADDX><<<dim3(D / 128, B / 128), blk, 0, stream>>>(
        h2, W3T, b3, nullptr, x, z, nullptr, nullptr, B, D, H);

    // Power-series: w <- J^T w; logdet += (-1)^{k+1}/k * <w, eps> per row.
    __bf16* t2 = h1;   // forward activations no longer needed: alias buffers
    __bf16* t1 = h2;
    for (int k = 1; k <= 8; ++k) {
        gemm_wmma_bf16<EPI_MULD><<<dim3(H / 128, B / 128), blk, 0, stream>>>(
            wv, W3b, nullptr, d2, nullptr, nullptr, t2, nullptr, B, H, D);
        gemm_wmma_bf16<EPI_MULD><<<dim3(H / 128, B / 128), blk, 0, stream>>>(
            t2, W2b, nullptr, d1, nullptr, nullptr, t1, nullptr, B, H, H);
        gemm_wmma_bf16<EPI_PLAIN><<<dim3(D / 128, B / 128), blk, 0, stream>>>(
            t1, W1b, nullptr, nullptr, nullptr, nullptr, wv, nullptr, B, D, H);
        float coeff = ((k & 1) ? 1.0f : -1.0f) / (float)k;
        rowdot_acc<<<B / 8, 256, 0, stream>>>(wv, eps, logdet, coeff, D);
    }
}
